// SelfAttention_61186104099586
// MI455X (gfx1250) — matrix operand order repaired and checked by reading, not compiled, not run on hardware
//
#include <hip/hip_runtime.h>
#include <hip/hip_bf16.h>

typedef __attribute__((ext_vector_type(16))) __bf16       v16bf;
typedef __attribute__((ext_vector_type(8)))  float        v8f;
typedef __attribute__((ext_vector_type(4)))  unsigned int v4u;

union BFrag { v4u u[2]; v16bf v; };

__device__ __forceinline__ unsigned short f2bf(float x) {
    union { float f; unsigned u; } c; c.f = x;
    unsigned r = c.u + 0x7fffu + ((c.u >> 16) & 1u);   // round-nearest-even
    return (unsigned short)(r >> 16);
}

__device__ __forceinline__ float sigmoid_(float x) {
    return 1.0f / (1.0f + __expf(-x));
}

__device__ __forceinline__ v8f wmma_bf16(const BFrag& a, const BFrag& b, v8f c) {
    // D = A(16x32 bf16) * B(32x16 bf16) + C(16x16 f32)
    return __builtin_amdgcn_wmma_f32_16x16x32_bf16(
        false, a.v, false, b.v, (short)0, c, false, false);
}

// async copies global -> LDS (ASYNCcnt); INST_OFFSET applies to BOTH addresses
__device__ __forceinline__ void async_b128_x2(unsigned lds, unsigned long long gaddr) {
    asm volatile("global_load_async_to_lds_b128 %0, %1, off" :: "v"(lds), "v"(gaddr) : "memory");
    asm volatile("global_load_async_to_lds_b128 %0, %1, off offset:64" :: "v"(lds), "v"(gaddr) : "memory");
}
__device__ __forceinline__ void async_b128(unsigned lds, unsigned long long gaddr) {
    asm volatile("global_load_async_to_lds_b128 %0, %1, off" :: "v"(lds), "v"(gaddr) : "memory");
}

// ---------------------------------------------------------------------------
// Prep: Q,K  f32 [n][s][h*64+d]  ->  bf16 [nh][s][64]
// ---------------------------------------------------------------------------
__global__ void prep_qk(const float* __restrict__ q, const float* __restrict__ k,
                        unsigned short* __restrict__ Qb, unsigned short* __restrict__ Kb) {
    unsigned t  = blockIdx.x * 256u + threadIdx.x;      // 4M threads
    unsigned d  = t & 63u;
    unsigned s  = (t >> 6) & 1023u;
    unsigned nh = t >> 16;
    unsigned n  = nh >> 4, h = nh & 15u;
    size_t src = ((size_t)(n * 1024u + s)) * 1024u + h * 64u + d;
    Qb[t] = f2bf(q[src]);
    Kb[t] = f2bf(k[src]);
}

// ---------------------------------------------------------------------------
// Prep: V  f32 [n][s][h*64+d]  ->  bf16 transposed [nh][64][s]  (LDS tiled)
// ---------------------------------------------------------------------------
__global__ void prep_v(const float* __restrict__ v, unsigned short* __restrict__ Vt) {
    __shared__ unsigned short tile[64][65];             // skewed to dodge bank conflicts
    unsigned nh = blockIdx.x >> 4;                      // 64 heads
    unsigned s0 = (blockIdx.x & 15u) * 64u;             // 16 s-tiles of 64
    unsigned n  = nh >> 4, h = nh & 15u;
    unsigned t  = threadIdx.x;
    unsigned d  = t & 63u, r = t >> 6;                  // r in 0..3
#pragma unroll
    for (unsigned p = 0; p < 16; ++p) {
        unsigned sl = p * 4u + r;
        tile[d][sl] = f2bf(v[((size_t)(n * 1024u + s0 + sl)) * 1024u + h * 64u + d]);
    }
    __syncthreads();
    unsigned sc = t & 63u, rr = t >> 6;
#pragma unroll
    for (unsigned p = 0; p < 16; ++p) {
        unsigned dl = p * 4u + rr;
        Vt[((size_t)(nh * 64u + dl)) * 1024u + s0 + sc] = tile[dl][sc];
    }
}

// ---------------------------------------------------------------------------
// Prep: W_fc f32 -> bf16 ; mask int32 -> u8
// ---------------------------------------------------------------------------
__global__ void prep_w(const float* __restrict__ W, unsigned short* __restrict__ Wb) {
    unsigned t = blockIdx.x * 256u + threadIdx.x;       // 1M threads
    Wb[t] = f2bf(W[t]);
}

__global__ void prep_m(const int* __restrict__ m, unsigned char* __restrict__ Mb) {
    unsigned t = blockIdx.x * 256u + threadIdx.x;       // 4M threads
    Mb[t] = (unsigned char)(m[t] != 0);
}

// ---------------------------------------------------------------------------
// Fused sigmoid-attention.  One wave = one 16-row q strip of one (n,h).
// Block = 8 waves = 128 q rows of one head.  64-key blocks staged into LDS
// with a quad-buffered async pipeline: one barrier / 16 WMMAs per step.
// Grid = 64 heads * 8 q-blocks.
// ---------------------------------------------------------------------------
__global__ void __launch_bounds__(256, 1) attn_kernel(
    const unsigned short* __restrict__ Qb, const unsigned short* __restrict__ Kb,
    const unsigned short* __restrict__ Vt, const unsigned char* __restrict__ Mb,
    unsigned short* __restrict__ X) {
    __shared__ __align__(16) unsigned short Kt[4][64 * 64];    // 4 x 8 KB (keys x d)
    __shared__ __align__(16) unsigned short Vtl[4][64 * 64];   // 4 x 8 KB (d x keys)
    __shared__ __align__(16) unsigned short Pbuf[8][16 * 64];  // 2 KB per wave

    const unsigned tid  = threadIdx.x;
    const unsigned lane = tid & 31u;
    const unsigned wave = tid >> 5;
    const unsigned ln   = lane & 15u;      // row (A) / col (B,C,D) within tile
    const unsigned hh   = lane >> 4;       // lane-half selector
    const unsigned nh   = blockIdx.x >> 3;
    const unsigned n    = nh >> 4, h = nh & 15u;
    const unsigned qg   = (blockIdx.x & 7u) * 128u + wave * 16u;

    // staging roles: both tiles are 64 rows x 128B; thread covers chunks c, c+4
    const unsigned srow = tid >> 2, schk = tid & 3u;
    const unsigned short* kgp = Kb + ((size_t)(nh * 1024u + srow)) * 64u + schk * 8u;
    const unsigned short* vgp = Vt + ((size_t)(nh * 64u + srow)) * 1024u + schk * 8u;

    auto stage = [&](unsigned kb, unsigned buf) {
        unsigned kl = (unsigned)(size_t)&Kt[buf][srow * 64u + schk * 8u];
        async_b128_x2(kl, (unsigned long long)(size_t)(kgp + (size_t)kb * 64u));
        unsigned vl = (unsigned)(size_t)&Vtl[buf][srow * 64u + schk * 8u];
        async_b128_x2(vl, (unsigned long long)(size_t)(vgp + kb));
    };

    // Q A-fragments (d-chunks 0..31, 32..63); row = 64 bf16 = 8 v4u
    const v4u* qp = (const v4u*)(Qb + ((size_t)(nh * 1024u + qg + ln)) * 64u);
    BFrag aq0, aq1;
    aq0.u[0] = qp[hh];      aq0.u[1] = qp[2u + hh];     // K = hh*8, 16+hh*8
    aq1.u[0] = qp[4u + hh]; aq1.u[1] = qp[6u + hh];     // K = 32+hh*8, 48+hh*8

    v8f o[4];
#pragma unroll
    for (int t = 0; t < 4; ++t) o[t] = v8f{0, 0, 0, 0, 0, 0, 0, 0};

    const unsigned char* mrow = Mb + (size_t)n * 1024u * 1024u;
    unsigned short* pw = &Pbuf[wave][0];

    stage(0u, 0u);                                     // prologue fill of buf 0
    for (unsigned it = 0; it < 16u; ++it) {
        const unsigned kb  = it * 64u;
        const unsigned buf = it & 3u;
        if (it + 1u < 16u) {
            stage(kb + 64u, (it + 1u) & 3u);           // prefetch next key block
            asm volatile("s_wait_asynccnt 0x4" ::: "memory");  // current buf done
        } else {
            asm volatile("s_wait_asynccnt 0x0" ::: "memory");
        }
        __syncthreads();

        const unsigned short* ktb = &Kt[buf][0];
        const unsigned short* vtb = &Vtl[buf][0];

        // ---- S = Q * K^T : batch-load all 8 K B-frags, then 8 WMMAs ----
        BFrag kf[4][2];
#pragma unroll
        for (int t2 = 0; t2 < 4; ++t2) {
            const v4u* kp = (const v4u*)(ktb + ((unsigned)t2 * 16u + ln) * 64u);
            kf[t2][0].u[0] = kp[hh];      kf[t2][0].u[1] = kp[2u + hh];      // d 0..31
            kf[t2][1].u[0] = kp[4u + hh]; kf[t2][1].u[1] = kp[6u + hh]; // d 32..63
        }
        v8f s[4];
#pragma unroll
        for (int t2 = 0; t2 < 4; ++t2) {
            v8f sv = v8f{0, 0, 0, 0, 0, 0, 0, 0};
            sv = wmma_bf16(aq0, kf[t2][0], sv);
            sv = wmma_bf16(aq1, kf[t2][1], sv);
            s[t2] = sv;
        }

        // ---- mask + sigmoid gate, pack P (16x64 bf16) to per-wave LDS ----
#pragma unroll
        for (int v = 0; v < 8; ++v) {
            unsigned qr = (unsigned)v + 8u * hh;        // C/D row for this VGPR
            const unsigned char* mq = mrow + (size_t)(qg + qr) * 1024u + kb;
#pragma unroll
            for (int t2 = 0; t2 < 4; ++t2) {
                float p = mq[(unsigned)t2 * 16u + ln]
                              ? sigmoid_(s[t2][v] * 0.03125f) : 0.0f;
                pw[qr * 64u + (unsigned)t2 * 16u + ln] = f2bf(p);
            }
        }
        asm volatile("s_wait_dscnt 0x0" ::: "memory");  // per-wave LDS RAW fence
        __builtin_amdgcn_wave_barrier();

        // reload P in A-fragment layout (2 chunks of 32 keys)
        BFrag pa0, pa1;
        const v4u* pp = (const v4u*)(pw + ln * 64u);
        pa0.u[0] = pp[hh];      pa0.u[1] = pp[2u + hh];
        pa1.u[0] = pp[4u + hh]; pa1.u[1] = pp[6u + hh];

        // ---- O += P * V : batch-load all 8 V B-frags, then 8 WMMAs ----
        BFrag bv[4][2];
#pragma unroll
        for (int t2 = 0; t2 < 4; ++t2) {
            const v4u* vp = (const v4u*)(vtb + ((unsigned)t2 * 16u + ln) * 64u);
            bv[t2][0].u[0] = vp[hh];      bv[t2][0].u[1] = vp[2u + hh];      // keys 0..31
            bv[t2][1].u[0] = vp[4u + hh]; bv[t2][1].u[1] = vp[6u + hh]; // keys 32..63
        }
#pragma unroll
        for (int t2 = 0; t2 < 4; ++t2) {
            o[t2] = wmma_bf16(pa0, bv[t2][0], o[t2]);
            o[t2] = wmma_bf16(pa1, bv[t2][1], o[t2]);
        }
        // quad buffering: next overwrite of `buf` is 3 barriers away -> safe
    }

    // ---- store attention output as bf16 X[n*1024+q][h*64+d] ----
#pragma unroll
    for (int t = 0; t < 4; ++t) {
#pragma unroll
        for (int v = 0; v < 8; ++v) {
            unsigned qr = qg + (unsigned)v + 8u * hh;
            X[(size_t)(n * 1024u + qr) * 1024u + h * 64u + (unsigned)t * 16u + ln] = f2bf(o[t][v]);
        }
    }
}

// ---------------------------------------------------------------------------
// FC: Y[4096][1024] = X * W^T + b.  One wave = 32 rows x 64 cols; the 8 waves
// of a block share the same 32 rows -> X tile quad-buffered via async copy.
// 64-wide k-steps: 16 WMMAs per barrier.
// ---------------------------------------------------------------------------
__global__ void __launch_bounds__(256, 1) fc_kernel(
    const unsigned short* __restrict__ X, const unsigned short* __restrict__ Wb,
    const float* __restrict__ bias, float* __restrict__ Y) {
    __shared__ __align__(16) unsigned short Xt[4][32 * 64];   // 4 x 4 KB

    const unsigned tid  = threadIdx.x;
    const unsigned lane = tid & 31u;
    const unsigned wave = tid >> 5;
    const unsigned ln   = lane & 15u;
    const unsigned hh   = lane >> 4;
    const unsigned w    = blockIdx.x * 8u + wave;       // 2048 waves
    const unsigned r0   = (w >> 4) * 32u;               // shared by whole block
    const unsigned c0   = (w & 15u) * 64u;

    // staging roles: 32 rows x 8 16B-chunks = 4 KB, 1 async per thread
    const unsigned xrow = tid >> 3, xchk = tid & 7u;
    const unsigned short* xgp = X + (size_t)(r0 + xrow) * 1024u + xchk * 8u;

    auto stage = [&](unsigned kb, unsigned buf) {
        unsigned xl = (unsigned)(size_t)&Xt[buf][xrow * 64u + xchk * 8u];
        async_b128(xl, (unsigned long long)(size_t)(xgp + kb));
    };

    v8f acc[2][4];
#pragma unroll
    for (int i = 0; i < 2; ++i)
#pragma unroll
        for (int t = 0; t < 4; ++t) acc[i][t] = v8f{0, 0, 0, 0, 0, 0, 0, 0};

    stage(0u, 0u);
    for (unsigned it = 0; it < 16u; ++it) {
        const unsigned kb  = it * 64u;
        const unsigned buf = it & 3u;
        if (it + 1u < 16u) {
            stage(kb + 64u, (it + 1u) & 3u);
            asm volatile("s_wait_asynccnt 0x1" ::: "memory");
        } else {
            asm volatile("s_wait_asynccnt 0x0" ::: "memory");
        }
        __syncthreads();

        // A-frags: 2 row strips x 2 k-chunks from LDS (row = 64 bf16 = 8 v4u)
        BFrag a[2][2];
#pragma unroll
        for (int i = 0; i < 2; ++i) {
            const v4u* xp = (const v4u*)(&Xt[buf][0] + ((unsigned)i * 16u + ln) * 64u);
#pragma unroll
            for (int c = 0; c < 2; ++c) {
                a[i][c].u[0] = xp[4u * c + hh];          // k = kb + c*32 + hh*8
                a[i][c].u[1] = xp[4u * c + 2u + hh];     // k = kb + c*32 + 16 + hh*8
            }
        }
        // B-frags from global (W rows are wave-private; L2-resident): batch 8
        BFrag b[4][2];
#pragma unroll
        for (int t = 0; t < 4; ++t) {
            const v4u* wp = (const v4u*)(Wb + (size_t)(c0 + (unsigned)t * 16u + ln) * 1024u + kb);
#pragma unroll
            for (int c = 0; c < 2; ++c) {
                b[t][c].u[0] = wp[4u * c + hh];
                b[t][c].u[1] = wp[4u * c + 2u + hh];
            }
        }
#pragma unroll
        for (int t = 0; t < 4; ++t)
#pragma unroll
            for (int c = 0; c < 2; ++c)
#pragma unroll
                for (int i = 0; i < 2; ++i)
                    acc[i][t] = wmma_bf16(a[i][c], b[t][c], acc[i][t]);
    }

#pragma unroll
    for (int t = 0; t < 4; ++t) {
        float bv = bias[c0 + (unsigned)t * 16u + ln];
#pragma unroll
        for (int i = 0; i < 2; ++i)
#pragma unroll
            for (int v = 0; v < 8; ++v) {
                unsigned r = r0 + (unsigned)i * 16u + (unsigned)v + 8u * hh;
                Y[(size_t)r * 1024u + c0 + (unsigned)t * 16u + ln] = acc[i][t][v] + bv;
            }
    }
}

// ---------------------------------------------------------------------------
extern "C" void kernel_launch(void* const* d_in, const int* in_sizes, int n_in,
                              void* d_out, int out_size, void* d_ws, size_t ws_size,
                              hipStream_t stream) {
    const float* values  = (const float*)d_in[0];
    const float* keys    = (const float*)d_in[1];
    const float* queries = (const float*)d_in[2];
    const int*   mask    = (const int*)d_in[3];
    const float* W_fc    = (const float*)d_in[4];
    const float* b_fc    = (const float*)d_in[5];
    float* out = (float*)d_out;

    char* ws = (char*)d_ws;                    // layout (38 MB total):
    unsigned short* Qb = (unsigned short*)(ws);                //  0 ..  8 MB
    unsigned short* Kb = (unsigned short*)(ws + (8u  << 20));  //  8 .. 16 MB
    unsigned short* Vt = (unsigned short*)(ws + (16u << 20));  // 16 .. 24 MB
    unsigned short* X  = (unsigned short*)(ws + (24u << 20));  // 24 .. 32 MB
    unsigned short* Wb = (unsigned short*)(ws + (32u << 20));  // 32 .. 34 MB
    unsigned char*  Mb = (unsigned char*) (ws + (34u << 20));  // 34 .. 38 MB

    prep_qk    <<<16384, 256, 0, stream>>>(queries, keys, Qb, Kb);
    prep_v     <<<1024,  256, 0, stream>>>(values, Vt);
    prep_w     <<<4096,  256, 0, stream>>>(W_fc, Wb);
    prep_m     <<<16384, 256, 0, stream>>>(mask, Mb);
    attn_kernel<<<512,   256, 0, stream>>>(Qb, Kb, Vt, Mb, X);
    fc_kernel  <<<256,   256, 0, stream>>>(X, Wb, b_fc, out);
}
